// ScaledDotProduct_71012989272727
// MI455X (gfx1250) — matrix operand order repaired and checked by reading, not compiled, not run on hardware
//
#include <hip/hip_runtime.h>
#include <hip/hip_bf16.h>

// SDPA (masked, thresholded softmax) for B=4,H=16,S=2048,D=64, fp32 I/O.
// Flash-attention over 32-key tiles using v_wmma_f32_16x16x32_f16, with
// double-buffered LDS staging (stage tile kt+1 while computing tile kt,
// single workgroup barrier per iteration).

typedef __attribute__((ext_vector_type(16))) _Float16 v16h;
typedef __attribute__((ext_vector_type(8)))  _Float16 v8h;
typedef __attribute__((ext_vector_type(2)))  _Float16 v2h;
typedef __attribute__((ext_vector_type(8)))  float    v8f;
typedef __attribute__((ext_vector_type(4)))  float    v4f;

#define S_LEN   2048
#define D_DIM   64
#define KT      32              // keys per iteration
#define NIT     (S_LEN / KT)    // 64 iterations
#define KPITCH  72              // halfwords per Ks row (64 + 8 pad, 144B = 9*16)
#define VPITCH  40              // halfwords per Vt row (32 + 8 pad, 80B = 5*16)
#define PPITCH  40              // halfwords per P row

__device__ __forceinline__ v16h cat16(v8h lo, v8h hi) {
  return __builtin_shufflevector(lo, hi, 0,1,2,3,4,5,6,7,8,9,10,11,12,13,14,15);
}

__device__ __forceinline__ v8f wmma_f16(v16h a, v16h b, v8f c) {
  // D = A*B + C, f16 inputs, f32 accumulate (16x16x32)
  return __builtin_amdgcn_wmma_f32_16x16x32_f16(false, a, false, b,
                                                (short)0, c, false, false);
}

// Cooperative staging by all 128 threads: K tile (32x64 f32) -> row-major f16,
// V tile (32x64 f32) -> transposed f16 Vt[d][k]. Prefetches tile kt+1 ahead.
__device__ __forceinline__ void stage_kv(_Float16* __restrict__ Ksb,
                                         _Float16* __restrict__ Vtb,
                                         const float* __restrict__ Kp,
                                         const float* __restrict__ Vp,
                                         int kbase, int pfbase, int tid) {
  #pragma unroll
  for (int it = 0; it < 2; ++it) {
    const int i  = tid + it * 128;
    const int r  = i >> 3;              // 0..31
    const int cs = (i & 7) * 8;         // 0..56
    const float* src = Kp + (size_t)(kbase + r) * D_DIM + cs;
    __builtin_prefetch(Kp + (size_t)(pfbase + r) * D_DIM + cs, 0, 1);
    v4f x = *(const v4f*)(src);
    v4f y = *(const v4f*)(src + 4);
    v8h o8;
    #pragma unroll
    for (int e = 0; e < 4; ++e) {
      o8[e]     = (_Float16)x[e];
      o8[4 + e] = (_Float16)y[e];
    }
    *(v8h*)&Ksb[r * KPITCH + cs] = o8;
  }
  #pragma unroll
  for (int it = 0; it < 2; ++it) {
    const int i  = tid + it * 128;
    const int rp = i >> 4;              // 0..15 (row pair)
    const int cq = (i & 15) * 4;        // 0..60
    const float* s0 = Vp + (size_t)(kbase + 2 * rp) * D_DIM + cq;
    __builtin_prefetch(Vp + (size_t)(pfbase + 2 * rp) * D_DIM + cq, 0, 1);
    v4f xa = *(const v4f*)(s0);
    v4f xb = *(const v4f*)(s0 + D_DIM);
    #pragma unroll
    for (int e = 0; e < 4; ++e) {
      v2h w;
      w[0] = (_Float16)xa[e];
      w[1] = (_Float16)xb[e];
      *(v2h*)&Vtb[(cq + e) * VPITCH + 2 * rp] = w;
    }
  }
}

__global__ __launch_bounds__(128)
void sdpa_flash_wmma(const float* __restrict__ Q, const float* __restrict__ K,
                     const float* __restrict__ V, const int* __restrict__ M,
                     float* __restrict__ O) {
  // Double-buffered K/V staging + per-wave P patch.  ~24.6 KB of 320 KB/WGP.
  __shared__ __align__(16) _Float16 Ks[2][32 * KPITCH];
  __shared__ __align__(16) _Float16 Vt[2][64 * VPITCH];
  __shared__ __align__(16) _Float16 Ps[4][16 * PPITCH];

  const int blk  = blockIdx.x;      // B*H*(S/64) = 2048 blocks
  const int qblk = blk & 31;        // 32 q-blocks of 64 rows
  const int bh   = blk >> 5;        // 0..63
  const int b    = bh >> 4;         // batch (mask is broadcast over heads)

  const float* Qp = Q + (size_t)bh * S_LEN * D_DIM;
  const float* Kp = K + (size_t)bh * S_LEN * D_DIM;
  const float* Vp = V + (size_t)bh * S_LEN * D_DIM;
  const int*   Mp = M + (size_t)b  * S_LEN * S_LEN;
  float*       Op = O + (size_t)bh * S_LEN * D_DIM;

  const int tid  = threadIdx.x;
  const int wave = tid >> 5;
  const int lane = tid & 31;
  const int hlf  = lane >> 4;       // lane half (WMMA layout)
  const int mrow = lane & 15;
  const int q0   = qblk * 64 + wave * 16;

  // ---------- Load Q tile into A-operand layout (f32 -> f16) ----------
  // A 16x32 f16 per chunk: lane holds row (lane%16), K = 8*hlf + t and 16 + 8*hlf + t.
  v16h aQ[2];
  {
    const float* qr = Qp + (size_t)(q0 + mrow) * D_DIM;
    #pragma unroll
    for (int c = 0; c < 2; ++c) {
      const float* p0 = qr + 32 * c + 8 * hlf;
      const float* p1 = qr + 32 * c + 16 + 8 * hlf;
      v4f x0 = *(const v4f*)(p0);
      v4f x1 = *(const v4f*)(p0 + 4);
      v4f y0 = *(const v4f*)(p1);
      v4f y1 = *(const v4f*)(p1 + 4);
      v16h a;
      #pragma unroll
      for (int e = 0; e < 4; ++e) {
        a[e]      = (_Float16)x0[e];
        a[4 + e]  = (_Float16)x1[e];
        a[8 + e]  = (_Float16)y0[e];
        a[12 + e] = (_Float16)y1[e];
      }
      aQ[c] = a;
    }
  }

  // Output accumulators (4 d-tiles of 16 cols) + online-softmax state.
  v8f o[4] = {};
  float mst[8], lst[8];
  #pragma unroll
  for (int r = 0; r < 8; ++r) { mst[r] = -__builtin_inff(); lst[r] = 0.0f; }

  _Float16* myP = &Ps[wave][0];

  // Prologue: stage tile 0 into buffer 0.
  stage_kv(&Ks[0][0], &Vt[0][0], Kp, Vp, 0, KT, tid);
  __syncthreads();

  for (int kt = 0; kt < NIT; ++kt) {
    const int cur   = kt & 1;
    const int kbase = kt * KT;

    // ---------- Pipeline: issue global loads for tile kt+1 now ----------
    if (kt + 1 < NIT) {
      const int pf = (kt + 2 < NIT) ? (kt + 2) * KT : (kt + 1) * KT;
      stage_kv(&Ks[cur ^ 1][0], &Vt[cur ^ 1][0], Kp, Vp, kbase + KT, pf, tid);
    }

    // ---------- S = Q K^T for two 16-key sub-tiles ----------
    // B operand: lane holds key row (nt*16 + lane%16), d = 32c + 8*hlf + t and 32c + 16 + 8*hlf + t.
    const _Float16* Ksb = &Ks[cur][0];
    const _Float16* Vtb = &Vt[cur][0];
    v16h bK[2][2];
    #pragma unroll
    for (int nt = 0; nt < 2; ++nt) {
      #pragma unroll
      for (int c = 0; c < 2; ++c) {
        const _Float16* p = &Ksb[(nt * 16 + mrow) * KPITCH + 32 * c + 8 * hlf];
        bK[nt][c] = cat16(*(const v8h*)p, *(const v8h*)(p + 16));
      }
    }
    v8f acc0 = {}, acc1 = {};
    acc0 = wmma_f16(aQ[0], bK[0][0], acc0);
    acc0 = wmma_f16(aQ[1], bK[0][1], acc0);
    acc1 = wmma_f16(aQ[0], bK[1][0], acc1);
    acc1 = wmma_f16(aQ[1], bK[1][1], acc1);

    // ---------- scale, mask, threshold; tile row-max ----------
    float tmax[8];
    #pragma unroll
    for (int r = 0; r < 8; ++r) {
      const size_t moff = (size_t)(q0 + r + 8 * hlf) * S_LEN + kbase + mrow;
      const int m0 = Mp[moff];
      const int m1 = Mp[moff + 16];
      float s0 = acc0[r] * 0.125f;
      float s1 = acc1[r] * 0.125f;
      s0 = (m0 != 0 && s0 > 0.0f) ? s0 : -10000.0f;
      s1 = (m1 != 0 && s1 > 0.0f) ? s1 : -10000.0f;
      acc0[r] = s0;
      acc1[r] = s1;
      tmax[r] = fmaxf(s0, s1);
    }
    #pragma unroll
    for (int r = 0; r < 8; ++r) {
      float t = tmax[r];
      t = fmaxf(t, __shfl_xor(t, 1, 16));
      t = fmaxf(t, __shfl_xor(t, 2, 16));
      t = fmaxf(t, __shfl_xor(t, 4, 16));
      t = fmaxf(t, __shfl_xor(t, 8, 16));
      tmax[r] = t;
    }

    // ---------- online softmax update + P store (D-layout -> LDS) ----------
    #pragma unroll
    for (int r = 0; r < 8; ++r) {
      const float mn = fmaxf(mst[r], tmax[r]);
      const float al = __expf(mst[r] - mn);
      const float p0 = __expf(acc0[r] - mn);
      const float p1 = __expf(acc1[r] - mn);
      float ts = p0 + p1;
      ts += __shfl_xor(ts, 1, 16);
      ts += __shfl_xor(ts, 2, 16);
      ts += __shfl_xor(ts, 4, 16);
      ts += __shfl_xor(ts, 8, 16);
      mst[r] = mn;
      lst[r] = lst[r] * al + ts;
      #pragma unroll
      for (int j = 0; j < 4; ++j) o[j][r] *= al;
      myP[(r + 8 * hlf) * PPITCH + mrow]      = (_Float16)p0;
      myP[(r + 8 * hlf) * PPITCH + 16 + mrow] = (_Float16)p1;
    }

    // ---------- O += P V : P in A-layout (per-wave LDS, in-order), V^T in B-layout ----------
    v16h pA;
    {
      const _Float16* p0 = &myP[mrow * PPITCH + 8 * hlf];        // k = 8*hlf + 0..7
      const _Float16* p1 = &myP[mrow * PPITCH + 16 + 8 * hlf];   // k = 16 + 8*hlf + 0..7
      pA = cat16(*(const v8h*)p0, *(const v8h*)p1);
    }
    #pragma unroll
    for (int j = 0; j < 4; ++j) {
      const _Float16* p = &Vtb[(16 * j + mrow) * VPITCH + 8 * hlf];
      v16h vB = cat16(*(const v8h*)p, *(const v8h*)(p + 16));
      o[j] = wmma_f16(pA, vB, o[j]);
    }

    // Single barrier: staging of buffer cur^1 complete for everyone, and all
    // reads of buffer cur are done before it is re-staged next iteration.
    __syncthreads();
  }

  // ---------- normalize and write out ----------
  #pragma unroll
  for (int r = 0; r < 8; ++r) {
    const float inv = 1.0f / lst[r];
    const size_t ro = (size_t)(q0 + r + 8 * hlf) * D_DIM;
    #pragma unroll
    for (int j = 0; j < 4; ++j)
      Op[ro + 16 * j + mrow] = o[j][r] * inv;
  }
}

extern "C" void kernel_launch(void* const* d_in, const int* in_sizes, int n_in,
                              void* d_out, int out_size, void* d_ws, size_t ws_size,
                              hipStream_t stream) {
  (void)in_sizes; (void)n_in; (void)d_ws; (void)ws_size; (void)out_size;
  const float* q = (const float*)d_in[0];
  const float* k = (const float*)d_in[1];
  const float* v = (const float*)d_in[2];
  const int*   m = (const int*)d_in[3];
  // d_in[4] is d_k == 64, baked into the kernel as D_DIM.
  dim3 grid(4 * 16 * (S_LEN / 64));   // 2048 workgroups
  dim3 block(128);                    // 4 wave32s
  hipLaunchKernelGGL(sdpa_flash_wmma, grid, block, 0, stream,
                     q, k, v, m, (float*)d_out);
}
